// Block_8074538516582
// MI455X (gfx1250) — matrix operand order repaired and checked by reading, not compiled, not run on hardware
//
#include <hip/hip_runtime.h>

// ---------------------------------------------------------------------------
// CDNA5 (gfx1250) wave32 WMMA implementation of the attention+MoE block.
// bf16 A/B matrices, f32 accumulation via V_WMMA_F32_16X16X32_BF16.
// Round 2: GEMM upgraded to BM=128/BN=128/BK=32, wave tile 32x64 (8 wmma per
// K-step), register-staged LDS double buffering + global_prefetch_b8.
// ---------------------------------------------------------------------------

typedef __bf16 v16bf __attribute__((ext_vector_type(16)));
typedef float  v8f   __attribute__((ext_vector_type(8)));
typedef unsigned int   u32x4 __attribute__((ext_vector_type(4)));
typedef float          f32x4 __attribute__((ext_vector_type(4)));

union Frag {                 // one WMMA A/B operand (16 bf16 per lane)
    u32x4 u[2];
    v16bf v;
    unsigned short s[16];
};

// constants
#define TOK   16384          // B*T tokens
#define CH    192            // C
#define F4    768            // 4C
#define F2    384            // 2C
#define NB    32             // batch
#define NT    512            // seq len
#define NH    6              // heads
#define HD    32             // head dim
#define NBH   (NB*NH)

__device__ __forceinline__ unsigned short f2bf(float f) {
    unsigned int u = __float_as_uint(f);
    u += 0x7FFFu + ((u >> 16) & 1u);          // round-to-nearest-even
    return (unsigned short)(u >> 16);
}
__device__ __forceinline__ float silu_f(float x) { return x / (1.f + __expf(-x)); }
__device__ __forceinline__ float gelu_f(float x) { return 0.5f * x * (1.f + erff(x * 0.70710678118654752f)); }

__device__ __forceinline__ float wsum32(float v) {
#pragma unroll
    for (int m = 1; m < 32; m <<= 1) v += __shfl_xor(v, m, 32);
    return v;
}
__device__ __forceinline__ float wmax16(float v) {
#pragma unroll
    for (int m = 1; m < 16; m <<= 1) v = fmaxf(v, __shfl_xor(v, m, 32));
    return v;
}
__device__ __forceinline__ float wsum16(float v) {
#pragma unroll
    for (int m = 1; m < 16; m <<= 1) v += __shfl_xor(v, m, 32);
    return v;
}

__device__ __forceinline__ v8f wmma_bf16(const Frag& a, const Frag& b, v8f c) {
    return __builtin_amdgcn_wmma_f32_16x16x32_bf16(false, a.v, false, b.v, (short)0, c, false, false);
}

// ---------------------------------------------------------------------------
// Generic tiled GEMM:  C[M,N] = epilogue(A[M,K](bf16) @ B[K,N](bf16))
// BM=128 BN=128 BK=32, 256 threads = 8 waves in 4x2; each wave a 32x64 tile
// (2 M-frags x 4 N-frags = 8 v_wmma per K-step). Register-staged double
// buffering overlaps global loads with WMMA compute.
// ACT: 0 none, 1 silu, 2 gelu.  OUTBF: bf16 store.  HASB: +bias. HASR: +res.
// ---------------------------------------------------------------------------
template <int ACT, int OUTBF, int HASB, int HASR>
__global__ void __launch_bounds__(256) wmma_gemm_kernel(
    const unsigned short* __restrict__ A, const unsigned short* __restrict__ B,
    const float* __restrict__ bias, const float* __restrict__ res,
    void* __restrict__ Cout, int M, int N, int K, int ldc, int cOff)
{
    __shared__ __align__(16) unsigned short As[128 * 32];   // row major
    __shared__ __align__(16) unsigned short Bs[128 * 32];   // transposed [n][k]

    const int tid = threadIdx.x;
    const int m0 = blockIdx.y * 128;
    const int n0 = blockIdx.x * 128;
    const int lane = tid & 31, laneLo = lane & 15, hi = lane >> 4;
    const int wv = tid >> 5;
    const int rm = (wv & 3) * 32;          // wave row base in tile
    const int cn = (wv >> 2) * 64;         // wave col base in tile

    v8f acc[2][4] = {};
    const int arow = tid >> 1, akc = (tid & 1) * 16;   // A: 16 halves/thread
    const bool fullN = (n0 + 128) <= N;

    u32x4 areg[2];
    unsigned short breg[2][8];

    auto stageA = [&](int k0) {
        const unsigned short* p = A + (long)(m0 + arow) * K + k0 + akc;
        areg[0] = *(const u32x4*)p;
        areg[1] = *(const u32x4*)(p + 8);
    };
    auto stageB = [&](int k0) {
#pragma unroll
        for (int c = 0; c < 2; ++c) {
            int cc = tid * 2 + c;
            int kk = cc >> 4, nch = (cc & 15) * 8;
            if (fullN) {
                *(u32x4*)breg[c] = *(const u32x4*)(B + (long)(k0 + kk) * N + n0 + nch);
            } else {
#pragma unroll
                for (int t2 = 0; t2 < 8; ++t2) {
                    int n = n0 + nch + t2; if (n > N - 1) n = N - 1;
                    breg[c][t2] = B[(long)(k0 + kk) * N + n];
                }
            }
        }
    };
    auto commit = [&]() {
        *(u32x4*)(As + arow * 32 + akc)     = areg[0];
        *(u32x4*)(As + arow * 32 + akc + 8) = areg[1];
#pragma unroll
        for (int c = 0; c < 2; ++c) {
            int cc = tid * 2 + c;
            int kk = cc >> 4, nch = (cc & 15) * 8;
#pragma unroll
            for (int t2 = 0; t2 < 8; ++t2) Bs[(nch + t2) * 32 + kk] = breg[c][t2];
        }
    };

    stageA(0); stageB(0);
    commit();
    __syncthreads();

    for (int k0 = 0; k0 < K; k0 += 32) {
        const bool more = (k0 + 32) < K;
        if (more) { stageA(k0 + 32); stageB(k0 + 32); }   // overlap w/ compute
        if (k0 + 64 < K)
            __builtin_prefetch(A + (long)(m0 + arow) * K + k0 + 64 + akc, 0, 1);

        Frag afr[2], bfr[4];
#pragma unroll
        for (int mt = 0; mt < 2; ++mt) {
            int base = (rm + mt * 16 + laneLo) * 32 + hi * 16;
            afr[mt].u[0] = *(const u32x4*)(As + base);
            afr[mt].u[1] = *(const u32x4*)(As + base + 8);
        }
#pragma unroll
        for (int nt = 0; nt < 4; ++nt) {
            int base = (cn + nt * 16 + laneLo) * 32 + hi * 16;
            bfr[nt].u[0] = *(const u32x4*)(Bs + base);
            bfr[nt].u[1] = *(const u32x4*)(Bs + base + 8);
        }
#pragma unroll
        for (int mt = 0; mt < 2; ++mt)
#pragma unroll
            for (int nt = 0; nt < 4; ++nt)
                acc[mt][nt] = wmma_bf16(afr[mt], bfr[nt], acc[mt][nt]);
        __syncthreads();
        if (more) { commit(); __syncthreads(); }
    }

    // ---- epilogue: C frag layout: VGPR j, lanes0-15 M=j, lanes16-31 M=j+8 ----
#pragma unroll
    for (int mt = 0; mt < 2; ++mt)
#pragma unroll
        for (int nt = 0; nt < 4; ++nt)
#pragma unroll
            for (int j = 0; j < 8; ++j) {
                int row = m0 + rm + mt * 16 + hi * 8 + j;
                int col = n0 + cn + nt * 16 + laneLo;
                if (col < N) {
                    float v = acc[mt][nt][j];
                    if (HASB) v += bias[col];
                    if (ACT == 1) v = silu_f(v);
                    if (ACT == 2) v = gelu_f(v);
                    long o = (long)row * ldc + cOff + col;
                    if (HASR) v += res[o];
                    if (OUTBF) ((unsigned short*)Cout)[o] = f2bf(v);
                    else       ((float*)Cout)[o] = v;
                }
            }
}

// ---------------------------------------------------------------------------
// Flash attention: grid (T/64, B*H), 128 threads = 4 waves x 16 query rows.
// S = (Q*scale) K^T, clip +-30, online softmax, O = P V, denom + 1e-10.
// ---------------------------------------------------------------------------
__global__ void __launch_bounds__(128) flash_attn_kernel(
    const unsigned short* __restrict__ qb, const unsigned short* __restrict__ kb,
    const unsigned short* __restrict__ vtb, unsigned short* __restrict__ attn_o)
{
    __shared__ __align__(16) float Pld[4][16][32];

    const int tid = threadIdx.x, wv = tid >> 5, lane = tid & 31;
    const int laneLo = lane & 15, hi = lane >> 4;
    const int bh = blockIdx.y, b = bh / NH, hh = bh % NH;
    const int qbase = blockIdx.x * 64 + wv * 16;

    const unsigned short* Q  = qb  + (long)bh * NT * HD;
    const unsigned short* Kp = kb  + (long)bh * NT * HD;
    const unsigned short* Vt = vtb + (long)bh * HD * NT;

    Frag qa;
    {
        const unsigned short* p = Q + (long)(qbase + laneLo) * HD + hi * 16;
        qa.u[0] = *(const u32x4*)p;
        qa.u[1] = *(const u32x4*)(p + 8);
    }

    float mrun[8], drun[8];
    v8f o0 = {}, o1 = {};
#pragma unroll
    for (int j = 0; j < 8; ++j) { mrun[j] = -1e30f; drun[j] = 0.f; }

    for (int kt = 0; kt < NT; kt += 32) {
        Frag kf0, kf1;
        {
            const unsigned short* p0 = Kp + (long)(kt + laneLo) * HD + hi * 16;
            kf0.u[0] = *(const u32x4*)p0;  kf0.u[1] = *(const u32x4*)(p0 + 8);
            const unsigned short* p1 = Kp + (long)(kt + 16 + laneLo) * HD + hi * 16;
            kf1.u[0] = *(const u32x4*)p1;  kf1.u[1] = *(const u32x4*)(p1 + 8);
        }
        v8f z = {};
        v8f s0 = wmma_bf16(qa, kf0, z);
        v8f s1 = wmma_bf16(qa, kf1, z);

#pragma unroll
        for (int j = 0; j < 8; ++j) {
            float a = fminf(fmaxf(s0[j], -30.f), 30.f);
            float c = fminf(fmaxf(s1[j], -30.f), 30.f);
            float mx = wmax16(fmaxf(a, c));
            float nm = fmaxf(mrun[j], mx);
            float alpha = __expf(mrun[j] - nm);
            float e0 = __expf(a - nm);
            float e1 = __expf(c - nm);
            drun[j] = drun[j] * alpha + wsum16(e0 + e1);
            mrun[j] = nm;
            o0[j] *= alpha;  o1[j] *= alpha;
            int row = j + hi * 8;
            Pld[wv][row][laneLo]      = e0;
            Pld[wv][row][laneLo + 16] = e1;
        }
        __syncthreads();

        Frag pa;                               // P as A-fragment (16x32 bf16)
        {
            const float* base = &Pld[wv][laneLo][hi * 16];
            f32x4 f0 = *(const f32x4*)(base);
            f32x4 f1 = *(const f32x4*)(base + 4);
            f32x4 f2 = *(const f32x4*)(base + 8);
            f32x4 f3 = *(const f32x4*)(base + 12);
#pragma unroll
            for (int i = 0; i < 4; ++i) {
                pa.s[i]      = f2bf(f0[i]);
                pa.s[4 + i]  = f2bf(f1[i]);
                pa.s[8 + i]  = f2bf(f2[i]);
                pa.s[12 + i] = f2bf(f3[i]);
            }
        }
        __syncthreads();

        Frag vf0, vf1;
        {
            const unsigned short* p0 = Vt + (long)laneLo * NT + kt + hi * 16;
            vf0.u[0] = *(const u32x4*)p0;  vf0.u[1] = *(const u32x4*)(p0 + 8);
            const unsigned short* p1 = Vt + (long)(16 + laneLo) * NT + kt + hi * 16;
            vf1.u[0] = *(const u32x4*)p1;  vf1.u[1] = *(const u32x4*)(p1 + 8);
        }
        o0 = wmma_bf16(pa, vf0, o0);
        o1 = wmma_bf16(pa, vf1, o1);
    }

#pragma unroll
    for (int j = 0; j < 8; ++j) {
        int row = qbase + j + hi * 8;
        float dn = 1.f / (drun[j] + 1e-10f);
        long o = ((long)b * NT + row) * CH + hh * HD;
        attn_o[o + laneLo]      = f2bf(o0[j] * dn);
        attn_o[o + 16 + laneLo] = f2bf(o1[j] * dn);
    }
}

// ---------------------------------------------------------------------------
// Elementwise / LN / router kernels (wave32 cross-lane reductions)
// ---------------------------------------------------------------------------
__global__ void __launch_bounds__(256) cvt_bf16_kernel(
    const float* __restrict__ in, unsigned short* __restrict__ out, int n)
{
    int i = blockIdx.x * 256 + threadIdx.x;
    if (i < n) out[i] = f2bf(in[i]);
}

__global__ void __launch_bounds__(256) copy_f32_kernel(
    const float* __restrict__ in, float* __restrict__ out, int n)
{
    int i = blockIdx.x * 256 + threadIdx.x;
    if (i < n) out[i] = in[i];
}

// LN1 + struct embed -> bf16  (one wave per token, 6 elems/lane)
__global__ void __launch_bounds__(256) ln1_embed_kernel(
    const float* __restrict__ x, const int* __restrict__ idx,
    const float* __restrict__ g, const float* __restrict__ b,
    const float* __restrict__ semb, unsigned short* __restrict__ out)
{
    int wv = threadIdx.x >> 5, lane = threadIdx.x & 31;
    long row = (long)blockIdx.x * 8 + wv;
    const float* xr = x + row * CH;
    float v[6], s = 0.f;
#pragma unroll
    for (int i = 0; i < 6; ++i) { v[i] = xr[lane + i * 32]; s += v[i]; }
    float mu = wsum32(s) * (1.f / CH);
    float q = 0.f;
#pragma unroll
    for (int i = 0; i < 6; ++i) { float d = v[i] - mu; q += d * d; }
    float rs = rsqrtf(wsum32(q) * (1.f / CH) + 1e-5f);
    int id = idx[row];
    int sid = (id >= 1 && id <= 3) ? id : 0;
#pragma unroll
    for (int i = 0; i < 6; ++i) {
        int c = lane + i * 32;
        out[row * CH + c] = f2bf((v[i] - mu) * rs * g[c] + b[c] + semb[sid * CH + c]);
    }
}

// LN2 -> f32 + bf16
__global__ void __launch_bounds__(256) ln2_kernel(
    const float* __restrict__ h, const float* __restrict__ g, const float* __restrict__ b,
    float* __restrict__ xb, unsigned short* __restrict__ xbb)
{
    int wv = threadIdx.x >> 5, lane = threadIdx.x & 31;
    long row = (long)blockIdx.x * 8 + wv;
    const float* r = h + row * CH;
    float v[6], s = 0.f;
#pragma unroll
    for (int i = 0; i < 6; ++i) { v[i] = r[lane + i * 32]; s += v[i]; }
    float mu = wsum32(s) * (1.f / CH);
    float q = 0.f;
#pragma unroll
    for (int i = 0; i < 6; ++i) { float d = v[i] - mu; q += d * d; }
    float rs = rsqrtf(wsum32(q) * (1.f / CH) + 1e-5f);
#pragma unroll
    for (int i = 0; i < 6; ++i) {
        int c = lane + i * 32;
        float y = (v[i] - mu) * rs * g[c] + b[c];
        xb[row * CH + c] = y;
        xbb[row * CH + c] = f2bf(y);
    }
}

// LN over 768 with optional post-SiLU -> bf16 (one wave per row, 24/lane)
template <int POST>
__global__ void __launch_bounds__(256) ln_act768_kernel(
    const float* __restrict__ in, const float* __restrict__ g, const float* __restrict__ b,
    unsigned short* __restrict__ out)
{
    int wv = threadIdx.x >> 5, lane = threadIdx.x & 31;
    long row = (long)blockIdx.x * 8 + wv;
    const float* r = in + row * F4;
    float v[24], s = 0.f;
#pragma unroll
    for (int i = 0; i < 24; ++i) { v[i] = r[lane + i * 32]; s += v[i]; }
    float mu = wsum32(s) * (1.f / F4);
    float q = 0.f;
#pragma unroll
    for (int i = 0; i < 24; ++i) { float d = v[i] - mu; q += d * d; }
    float rs = rsqrtf(wsum32(q) * (1.f / F4) + 1e-5f);
#pragma unroll
    for (int i = 0; i < 24; ++i) {
        int c = lane + i * 32;
        float y = (v[i] - mu) * rs * g[c] + b[c];
        if (POST == 1) y = silu_f(y);
        out[row * F4 + c] = f2bf(y);
    }
}

// out += gate_e * LN(xb + y)  (expert residual+LN epilogue)
__global__ void __launch_bounds__(256) combine_kernel(
    const float* __restrict__ xb, const float* __restrict__ y,
    const float* __restrict__ gates, int e,
    const float* __restrict__ g, const float* __restrict__ b,
    float* __restrict__ out)
{
    int wv = threadIdx.x >> 5, lane = threadIdx.x & 31;
    long row = (long)blockIdx.x * 8 + wv;
    float gate = gates[row * 6 + e];
    float v[6], s = 0.f;
#pragma unroll
    for (int i = 0; i < 6; ++i) {
        int c = lane + i * 32;
        v[i] = xb[row * CH + c] + y[row * CH + c];
        s += v[i];
    }
    float mu = wsum32(s) * (1.f / CH);
    float q = 0.f;
#pragma unroll
    for (int i = 0; i < 6; ++i) { float d = v[i] - mu; q += d * d; }
    float rs = rsqrtf(wsum32(q) * (1.f / CH) + 1e-5f);
#pragma unroll
    for (int i = 0; i < 6; ++i) {
        int c = lane + i * 32;
        out[row * CH + c] += gate * ((v[i] - mu) * rs * g[c] + b[c]);
    }
}

// qkv reshape: apply the triton q-row mask + scale, split heads, transpose V
__global__ void __launch_bounds__(256) qkv_reshape_kernel(
    const float* __restrict__ qkv, unsigned short* __restrict__ qb,
    unsigned short* __restrict__ kb, unsigned short* __restrict__ vtb)
{
    long i = (long)blockIdx.x * 256 + threadIdx.x;   // (b,t,h,d), d fastest
    int d = i & 31;
    long r = i >> 5;
    int hh = (int)(r % NH);
    long bt = r / NH;                                 // b*NT + t
    int t = (int)(bt & (NT - 1));
    int b = (int)(bt >> 9);
    const float* src = qkv + bt * (3 * CH) + hh * HD + d;
    float qv = src[0], kv = src[CH], vv = src[2 * CH];
    float mask = (((t & 63) % 3) != 0) ? 1.f : 0.f;
    qv *= 0.17677669529663689f * mask;                // 1/sqrt(32), mask quirk
    long bh = (long)b * NH + hh;
    qb[(bh * NT + t) * HD + d] = f2bf(qv);
    kb[(bh * NT + t) * HD + d] = f2bf(kv);
    vtb[(bh * HD + d) * NT + t] = f2bf(vv);
}

// pre_t[e][j] = type_emb[e%3] @ W1[C:,:] + b1   (6 x 768)
__global__ void __launch_bounds__(256) pre_t_kernel(
    const float* __restrict__ type_emb, const float* __restrict__ l1w,
    const float* __restrict__ l1b, float* __restrict__ pre_t)
{
    int i = blockIdx.x * 256 + threadIdx.x;
    if (i >= 6 * F4) return;
    int e = i / F4, j = i % F4;
    const float* tf = type_emb + (e % 3) * F2;
    float acc = l1b[j];
    for (int k = 0; k < F2; ++k) acc += tf[k] * l1w[(long)(CH + k) * F4 + j];
    pre_t[i] = acc;
}

// w2m[k] = mean_j l2w[k][j]
__global__ void __launch_bounds__(256) w2m_kernel(
    const float* __restrict__ l2w, float* __restrict__ w2m)
{
    int i = blockIdx.x * 256 + threadIdx.x;
    if (i >= F4) return;
    float s = 0.f;
#pragma unroll
    for (int j = 0; j < 6; ++j) s += l2w[i * 6 + j];
    w2m[i] = s * (1.f / 6.f);
}

// router gating: logits[e] = (gelu(pre_x + pre_t[e]) . w2m + bm)/temp; top2 softmax
__global__ void __launch_bounds__(256) router_gate_kernel(
    const float* __restrict__ pre_x, const float* __restrict__ pre_t,
    const float* __restrict__ w2m, const float* __restrict__ l2b,
    const float* __restrict__ temp_p, float* __restrict__ gates)
{
    __shared__ float sT[6 * F4];
    __shared__ float sM[F4];
    for (int i = threadIdx.x; i < 6 * F4; i += 256) sT[i] = pre_t[i];
    for (int i = threadIdx.x; i < F4; i += 256) sM[i] = w2m[i];
    __syncthreads();

    int wv = threadIdx.x >> 5, lane = threadIdx.x & 31;
    long row = (long)blockIdx.x * 8 + wv;
    const float* px = pre_x + row * F4;
    float pv[24];
#pragma unroll
    for (int i = 0; i < 24; ++i) pv[i] = px[lane + i * 32];

    float bm = (l2b[0] + l2b[1] + l2b[2] + l2b[3] + l2b[4] + l2b[5]) * (1.f / 6.f);
    float temp = fminf(fmaxf(temp_p[0], 0.05f), 3.f);
    float invT = 1.f / temp;

    float logit[6];
#pragma unroll
    for (int e = 0; e < 6; ++e) {
        float acc = 0.f;
#pragma unroll
        for (int i = 0; i < 24; ++i) {
            int j = lane + i * 32;
            acc += gelu_f(pv[i] + sT[e * F4 + j]) * sM[j];
        }
        logit[e] = (wsum32(acc) + bm) * invT;
    }
    if (lane == 0) {
        int i1 = 0; float v1 = logit[0];
#pragma unroll
        for (int e = 1; e < 6; ++e) if (logit[e] > v1) { v1 = logit[e]; i1 = e; }
        int i2 = -1; float v2 = -1e30f;
#pragma unroll
        for (int e = 0; e < 6; ++e) if (e != i1 && logit[e] > v2) { v2 = logit[e]; i2 = e; }
        float w2 = __expf(v2 - v1);
        float inv = 1.f / (1.f + w2);
#pragma unroll
        for (int e = 0; e < 6; ++e)
            gates[row * 6 + e] = (e == i1) ? inv : ((e == i2) ? w2 * inv : 0.f);
    }
}

// ---------------------------------------------------------------------------
// Host launcher
// ---------------------------------------------------------------------------
extern "C" void kernel_launch(void* const* d_in, const int* in_sizes, int n_in,
                              void* d_out, int out_size, void* d_ws, size_t ws_size,
                              hipStream_t stream)
{
    (void)in_sizes; (void)n_in; (void)out_size; (void)ws_size;
    const int M = TOK;

    // -------- input leaf order (jax pytree: dicts sorted by key) --------
    const float* x          = (const float*)d_in[0];
    const int*   idx        = (const int*)d_in[1];
    const float* aout_b     = (const float*)d_in[2];
    const float* aout_w     = (const float*)d_in[3];
    const float* semb       = (const float*)d_in[4];
    const float* w_qkv      = (const float*)d_in[5];
    int ei[6][12];
    int pi = 6;
    for (int e = 0; e < 6; ++e) {
        int t = e % 3;
        int n = (t == 0) ? 10 : (t == 1 ? 8 : 12);
        for (int j = 0; j < n; ++j) ei[e][j] = pi + j;
        pi += n;
    }
    const float* ln1_b = (const float*)d_in[pi + 0];
    const float* ln1_g = (const float*)d_in[pi + 1];
    const float* ln2_b = (const float*)d_in[pi + 2];
    const float* ln2_g = (const float*)d_in[pi + 3];
    const float* r_l1b = (const float*)d_in[pi + 4];
    const float* r_l1w = (const float*)d_in[pi + 5];
    const float* r_l2b = (const float*)d_in[pi + 6];
    const float* r_l2w = (const float*)d_in[pi + 7];
    const float* r_tmp = (const float*)d_in[pi + 8];
    const float* r_tem = (const float*)d_in[pi + 9];
    auto F = [&](int i) { return (const float*)d_in[i]; };

    // -------- workspace bump allocator --------
    char* wsp = (char*)d_ws;
    size_t cur = 0;
    auto alloc = [&](size_t bytes) -> void* {
        void* p = wsp + cur;
        cur += (bytes + 255) & ~(size_t)255;
        return p;
    };
    unsigned short* wq_bf   = (unsigned short*)alloc((size_t)CH * 3 * CH * 2);
    unsigned short* wout_bf = (unsigned short*)alloc((size_t)CH * CH * 2);
    unsigned short* w1x_bf  = (unsigned short*)alloc((size_t)CH * F4 * 2);
    unsigned short* ew[6][5];
    for (int e = 0; e < 6; ++e) {
        int t = e % 3;
        if (t == 0) {
            ew[e][0] = (unsigned short*)alloc((size_t)CH * F4 * 2);
            ew[e][1] = (unsigned short*)alloc((size_t)F4 * F4 * 2);
            ew[e][2] = (unsigned short*)alloc((size_t)F4 * CH * 2);
        } else if (t == 1) {
            ew[e][0] = (unsigned short*)alloc((size_t)CH * F4 * 2);
            ew[e][1] = (unsigned short*)alloc((size_t)F4 * CH * 2);
        } else {
            ew[e][0] = (unsigned short*)alloc((size_t)CH * F4 * 2);
            ew[e][1] = (unsigned short*)alloc((size_t)F4 * CH * 2);
            ew[e][2] = (unsigned short*)alloc((size_t)CH * F2 * 2);
            ew[e][3] = (unsigned short*)alloc((size_t)F2 * CH * 2);
            ew[e][4] = (unsigned short*)alloc((size_t)F2 * CH * 2);
        }
    }
    float* pre_t  = (float*)alloc(6 * F4 * 4);
    float* w2m    = (float*)alloc(F4 * 4);
    float* gates  = (float*)alloc((size_t)M * 6 * 4);
    float* hbuf   = (float*)alloc((size_t)M * CH * 4);
    float* xb     = (float*)alloc((size_t)M * CH * 4);
    unsigned short* xb_bf = (unsigned short*)alloc((size_t)M * CH * 2);
    float* S1 = (float*)alloc((size_t)M * F4 * 4);            // qkv / pre_x / mid f32
    unsigned short* S2 = (unsigned short*)alloc((size_t)M * F4 * 2);  // q,k,vT / hmid bf16
    float* S3 = (float*)alloc((size_t)M * CH * 4);            // attn_o bf16 / y f32
    unsigned short* S4 = (unsigned short*)alloc((size_t)M * F2 * 2);  // xa_bf / cat

    unsigned short* xa_bf  = S4;
    float*          qkv    = S1;
    unsigned short* q_bf   = S2;
    unsigned short* k_bf   = S2 + (size_t)M * CH;
    unsigned short* vt_bf  = S2 + (size_t)2 * M * CH;
    unsigned short* attn_o = (unsigned short*)S3;
    float*          pre_x  = S1;
    float*          midf   = S1;
    unsigned short* hmid   = S2;
    float*          ybuf   = S3;
    unsigned short* catb   = S4;

    auto cvt = [&](const float* src, unsigned short* dst, int n) {
        cvt_bf16_kernel<<<(n + 255) / 256, 256, 0, stream>>>(src, dst, n);
    };
    auto ggrid = [](int Mv, int Nv) { return dim3((Nv + 127) / 128, Mv / 128); };

    // -------- weight conversion to bf16 --------
    cvt(w_qkv, wq_bf, CH * 3 * CH);
    cvt(aout_w, wout_bf, CH * CH);
    cvt(r_l1w, w1x_bf, CH * F4);          // first C rows of router l1.w
    for (int e = 0; e < 6; ++e) {
        int t = e % 3;
        if (t == 0) {
            cvt(F(ei[e][1]), ew[e][0], CH * F4);
            cvt(F(ei[e][3]), ew[e][1], F4 * F4);
            cvt(F(ei[e][5]), ew[e][2], F4 * CH);
        } else if (t == 1) {
            cvt(F(ei[e][1]), ew[e][0], CH * F4);
            cvt(F(ei[e][3]), ew[e][1], F4 * CH);
        } else {
            cvt(F(ei[e][1]),  ew[e][0], CH * F4);
            cvt(F(ei[e][3]),  ew[e][1], F4 * CH);
            cvt(F(ei[e][5]),  ew[e][2], CH * F2);
            cvt(F(ei[e][7]),  ew[e][3], F2 * CH);
            cvt(F(ei[e][11]), ew[e][4], F2 * CH);
        }
    }

    // -------- phase 1: attention --------
    ln1_embed_kernel<<<M / 8, 256, 0, stream>>>(x, idx, ln1_g, ln1_b, semb, xa_bf);
    wmma_gemm_kernel<0, 0, 0, 0><<<ggrid(M, 3 * CH), 256, 0, stream>>>(
        xa_bf, wq_bf, nullptr, nullptr, qkv, M, 3 * CH, CH, 3 * CH, 0);
    qkv_reshape_kernel<<<(NB * NT * NH * HD) / 256, 256, 0, stream>>>(qkv, q_bf, k_bf, vt_bf);
    flash_attn_kernel<<<dim3(NT / 64, NBH), 128, 0, stream>>>(q_bf, k_bf, vt_bf, attn_o);
    wmma_gemm_kernel<0, 0, 1, 1><<<ggrid(M, CH), 256, 0, stream>>>(
        attn_o, wout_bf, aout_b, x, hbuf, M, CH, CH, CH, 0);

    // -------- phase 2: LN2 + router --------
    ln2_kernel<<<M / 8, 256, 0, stream>>>(hbuf, ln2_g, ln2_b, xb, xb_bf);
    pre_t_kernel<<<(6 * F4 + 255) / 256, 256, 0, stream>>>(r_tem, r_l1w, r_l1b, pre_t);
    w2m_kernel<<<(F4 + 255) / 256, 256, 0, stream>>>(r_l2w, w2m);
    wmma_gemm_kernel<0, 0, 0, 0><<<ggrid(M, F4), 256, 0, stream>>>(
        xb_bf, w1x_bf, nullptr, nullptr, pre_x, M, F4, CH, F4, 0);
    router_gate_kernel<<<M / 8, 256, 0, stream>>>(pre_x, pre_t, w2m, r_l2b, r_tmp, gates);

    // -------- phase 3: experts (dense, gated) --------
    copy_f32_kernel<<<(M * CH) / 256, 256, 0, stream>>>(hbuf, (float*)d_out, M * CH);
    for (int e = 0; e < 6; ++e) {
        int t = e % 3;
        if (t == 0) {                       // deep: silu(L1) -> silu(LN(L2)) -> L3
            wmma_gemm_kernel<1, 1, 1, 0><<<ggrid(M, F4), 256, 0, stream>>>(
                xb_bf, ew[e][0], F(ei[e][0]), nullptr, hmid, M, F4, CH, F4, 0);
            wmma_gemm_kernel<0, 0, 1, 0><<<ggrid(M, F4), 256, 0, stream>>>(
                hmid, ew[e][1], F(ei[e][2]), nullptr, midf, M, F4, F4, F4, 0);
            ln_act768_kernel<1><<<M / 8, 256, 0, stream>>>(midf, F(ei[e][9]), F(ei[e][8]), hmid);
            wmma_gemm_kernel<0, 0, 1, 0><<<ggrid(M, CH), 256, 0, stream>>>(
                hmid, ew[e][2], F(ei[e][4]), nullptr, ybuf, M, CH, F4, CH, 0);
            combine_kernel<<<M / 8, 256, 0, stream>>>(
                xb, ybuf, gates, e, F(ei[e][7]), F(ei[e][6]), (float*)d_out);
        } else if (t == 1) {                // wide: LN(gelu(L1)) -> L2
            wmma_gemm_kernel<2, 0, 1, 0><<<ggrid(M, F4), 256, 0, stream>>>(
                xb_bf, ew[e][0], F(ei[e][0]), nullptr, midf, M, F4, CH, F4, 0);
            ln_act768_kernel<0><<<M / 8, 256, 0, stream>>>(midf, F(ei[e][7]), F(ei[e][6]), hmid);
            wmma_gemm_kernel<0, 0, 1, 0><<<ggrid(M, CH), 256, 0, stream>>>(
                hmid, ew[e][1], F(ei[e][2]), nullptr, ybuf, M, CH, F4, CH, 0);
            combine_kernel<<<M / 8, 256, 0, stream>>>(
                xb, ybuf, gates, e, F(ei[e][5]), F(ei[e][4]), (float*)d_out);
        } else {                            // hybrid: two paths + proj
            wmma_gemm_kernel<2, 1, 1, 0><<<ggrid(M, F4), 256, 0, stream>>>(
                xb_bf, ew[e][0], F(ei[e][0]), nullptr, hmid, M, F4, CH, F4, 0);
            wmma_gemm_kernel<0, 1, 1, 0><<<ggrid(M, CH), 256, 0, stream>>>(
                hmid, ew[e][1], F(ei[e][2]), nullptr, catb, M, CH, F4, F2, 0);
            wmma_gemm_kernel<1, 1, 1, 0><<<ggrid(M, F2), 256, 0, stream>>>(
                xb_bf, ew[e][2], F(ei[e][4]), nullptr, hmid, M, F2, CH, F2, 0);
            wmma_gemm_kernel<0, 1, 1, 0><<<ggrid(M, CH), 256, 0, stream>>>(
                hmid, ew[e][3], F(ei[e][6]), nullptr, catb, M, CH, F2, F2, CH);
            wmma_gemm_kernel<0, 0, 1, 0><<<ggrid(M, CH), 256, 0, stream>>>(
                catb, ew[e][4], F(ei[e][10]), nullptr, ybuf, M, CH, F2, CH, 0);
            combine_kernel<<<M / 8, 256, 0, stream>>>(
                xb, ybuf, gates, e, F(ei[e][9]), F(ei[e][8]), (float*)d_out);
        }
    }
}
